// BroadcastResidualBlock_32014686224994
// MI455X (gfx1250) — matrix operand order repaired and checked by reading, not compiled, not run on hardware
//
#include <hip/hip_runtime.h>

// ---------------------------------------------------------------------------
// BroadcastResidualBlock for MI455X (gfx1250, wave32, WMMA + TDM).
//
// 41 GFLOP / ~400 MB HBM traffic -> ~20us memory bound @ 23.3 TB/s provided
// the math runs on v_wmma_f32_16x16x32_f16 and LDS/DS traffic stays under
// the matrix-pipe time. Weights are staged into a double-buffered LDS panel
// by the Tensor Data Mover (tensor_load_to_lds, TENSORcnt), overlapped with
// compute; each wave owns 32 M rows so every B fragment (2x ds_load_b128)
// feeds two WMMAs, halving DS bandwidth per FLOP.
// ---------------------------------------------------------------------------

typedef _Float16 h16;
typedef __attribute__((ext_vector_type(16))) _Float16 v16h;
typedef __attribute__((ext_vector_type(8)))  _Float16 v8h;
typedef __attribute__((ext_vector_type(8)))  float    v8f;

typedef __attribute__((ext_vector_type(4))) unsigned int tdm_u32x4;
typedef __attribute__((ext_vector_type(8))) int          tdm_i32x8;
typedef __attribute__((ext_vector_type(4))) int          tdm_i32x4;

#define NIMG 256
#define HW   361          // 19*19
#define HWP  384          // padded to 12 K-steps of 32 (3 chunks of 128)
#define CCH  256
#define NHW  (NIMG * HW)  // 92416 rows -> 361 M-blocks of 256
#define M2   (NIMG * CCH) // 65536 rows -> 256 M-blocks of 256
#define BN_EPS 1e-3f

#define PANEL (128 * 128) // halves per LDS B panel (32 KB)

__device__ __forceinline__ v16h mk16(v8h lo, v8h hi) {
    return __builtin_shufflevector(lo, hi, 0,1,2,3,4,5,6,7,8,9,10,11,12,13,14,15);
}

__device__ __forceinline__ v8f wmma_f16(v16h a, v16h b, v8f c) {
    return __builtin_amdgcn_wmma_f32_16x16x32_f16(false, a, false, b, (short)0, c,
                                                  false, false);
}

// A fragment, contiguous K order: lanes 0-15 -> K {0..15},
// lanes 16-31 -> K {16..31}; two 16B global loads from the lane's row.
__device__ __forceinline__ v16h load_a_frag(const h16* __restrict__ row, int k0, int lh) {
    v8h lo = *(const v8h*)(row + k0 + 16 * lh);
    v8h hi = *(const v8h*)(row + k0 + 16 * lh + 8);
    return mk16(lo, hi);
}

// ---------------------------------------------------------------------------
// TDM: DMA a [rows x cols] f16 tile (row stride in halves) from global into
// LDS. Descriptor layout per CDNA5 ISA;
// 6-arg builtin on this toolchain, groups 2/3 zero for a 2D tile.
// ---------------------------------------------------------------------------
__device__ __forceinline__ void tdm_load_tile(unsigned lds_byte_off, const h16* gptr,
                                              int rows, int cols, int row_stride)
{
    unsigned long long ga = (unsigned long long)(size_t)gptr;
    tdm_u32x4 g0;
    g0[0] = 1u;                                          // count=1, user desc
    g0[1] = lds_byte_off;                                // lds_addr
    g0[2] = (unsigned)(ga & 0xFFFFFFFFu);                // global_addr[31:0]
    g0[3] = (unsigned)((ga >> 32) & 0x1FFFFFFu)          // global_addr[56:32]
          | (2u << 30);                                  // type = 2 ("image")
    tdm_i32x8 g1;
    g1[0] = 1 << 16;                                     // data_size=1 -> 2 bytes
    g1[1] = (cols & 0xFFFF) << 16;                       // tensor_dim0[15:0]
    g1[2] = ((unsigned)cols >> 16) | ((rows & 0xFFFF) << 16);   // td0 hi | tensor_dim1 lo
    g1[3] = ((unsigned)rows >> 16) | ((cols & 0xFFFF) << 16);   // td1 hi | tile_dim0
    g1[4] = rows & 0xFFFF;                               // tile_dim1 (tile_dim2=0)
    g1[5] = row_stride;                                  // tensor_dim0_stride[31:0]
    g1[6] = 0;
    g1[7] = 0;
    tdm_i32x4 z4 = {};
    tdm_i32x8 z8 = {};
    __builtin_amdgcn_tensor_load_to_lds(g0, g1, z4, z4, z8, 0);
}

// ---------------------------------------------------------------------------
// GEMM core: block = 256 M rows x 128 N cols. Wave w owns M rows
// 32w..32w+31 (two A fragments); B panel [n][k] staged per 128-K chunk in
// double-buffered LDS by the TDM, prefetch of chunk ch+1 overlapped with
// compute of chunk ch.
// ---------------------------------------------------------------------------
template <int KCHUNKS>
__device__ __forceinline__ void gemm_core(const h16* __restrict__ arow0,
                                          const h16* __restrict__ arow1,
                                          const h16* __restrict__ wt,  // B, [n][k]
                                          int wstride, int nB, h16* lds,
                                          v8f acc0[8], v8f acc1[8],
                                          int lh, int r)
{
    const unsigned lds0 = (unsigned)(size_t)(void*)lds;   // LDS byte offset
    const bool issuer = threadIdx.x < 32;                 // wave 0 drives TDM
    if (issuer) {
        tdm_load_tile(lds0, wt + (size_t)nB * wstride, 128, 128, wstride);
        __builtin_amdgcn_s_wait_tensorcnt(0);
    }
    __syncthreads();

    for (int ch = 0; ch < KCHUNKS; ++ch) {
        if (issuer && ch + 1 < KCHUNKS)                   // async prefetch
            tdm_load_tile(lds0 + ((ch + 1) & 1) * (PANEL * 2),
                          wt + (size_t)nB * wstride + (ch + 1) * 128,
                          128, 128, wstride);
        const h16* bbuf = lds + (ch & 1) * PANEL;
        #pragma unroll
        for (int ks = 0; ks < 4; ++ks) {
            const int k0g = ch * 128 + ks * 32;
            __builtin_prefetch(arow0 + k0g + 32, 0, 3);
            __builtin_prefetch(arow1 + k0g + 32, 0, 3);
            v16h a0 = load_a_frag(arow0, k0g, lh);
            v16h a1 = load_a_frag(arow1, k0g, lh);
            #pragma unroll
            for (int t = 0; t < 8; ++t) {
                const h16* bp = bbuf + (t * 16 + r) * 128 + ks * 32 + 16 * lh;
                v16h b = mk16(*(const v8h*)bp, *(const v8h*)(bp + 8));
                acc0[t] = wmma_f16(a0, b, acc0[t]);       // B reused twice
                acc1[t] = wmma_f16(a1, b, acc1[t]);
            }
        }
        __syncthreads();                                  // buffer consumed
        if (issuer && ch + 1 < KCHUNKS)
            __builtin_amdgcn_s_wait_tensorcnt(0);         // next buffer ready
        __syncthreads();
    }
}

// ---------------------------------------------------------------------------
// prep: f32->f16 conversion, weight transposes, dense_w zero-padding, and
// zeroing of h1's K-padding columns (361..383).
// ---------------------------------------------------------------------------
__global__ __launch_bounds__(256)
void prep_kernel(const float* __restrict__ x,  const float* __restrict__ w1,
                 const float* __restrict__ w2, const float* __restrict__ wd,
                 h16* __restrict__ xh,  h16* __restrict__ w1t,
                 h16* __restrict__ w2t, h16* __restrict__ wdt,
                 h16* __restrict__ h1)
{
    long long i = (long long)blockIdx.x * blockDim.x + threadIdx.x;
    const long long TA = (long long)NHW * CCH;
    if (i < TA) { xh[i] = (h16)x[i]; return; }
    i -= TA;
    if (i < CCH * CCH) {
        int d = (int)(i >> 8), c = (int)(i & 255);
        w1t[i] = (h16)w1[c * CCH + d];
        return;
    }
    i -= CCH * CCH;
    if (i < CCH * CCH) {
        int d = (int)(i >> 8), c = (int)(i & 255);
        w2t[i] = (h16)w2[c * CCH + d];
        return;
    }
    i -= CCH * CCH;
    if (i < HWP * HWP) {
        int q = (int)(i / HWP), p = (int)(i % HWP);
        wdt[i] = (q < HW && p < HW) ? (h16)wd[(long long)p * HW + q] : (h16)0.f;
        return;
    }
    i -= (long long)HWP * HWP;
    if (i < (long long)M2 * (HWP - HW)) {
        int m = (int)(i / (HWP - HW));
        int p = HW + (int)(i % (HWP - HW));
        h1[(size_t)m * HWP + p] = (h16)0.f;
    }
}

// ---------------------------------------------------------------------------
// GEMM1: h1(n*C+d, p) = relu(BN(x(n*HW+p, c) @ w1(c,d) + b1))
// ---------------------------------------------------------------------------
__global__ __launch_bounds__(256)
void gemm1_kernel(const h16* __restrict__ xh, const h16* __restrict__ w1t,
                  const float* __restrict__ bias, const float* __restrict__ mean,
                  const float* __restrict__ var,  const float* __restrict__ beta,
                  h16* __restrict__ h1)
{
    __shared__ h16 lds[2 * PANEL];                 // 64 KB double-buffered B
    const int wv = threadIdx.x >> 5, lane = threadIdx.x & 31;
    const int lh = lane >> 4, r = lane & 15;
    const int mW = blockIdx.x * 256 + wv * 32;
    const int nB = blockIdx.y * 128;

    v8f acc0[8] = {}, acc1[8] = {};
    gemm_core<2>(xh + (size_t)(mW + r) * CCH, xh + (size_t)(mW + 16 + r) * CCH,
                 w1t, CCH, nB, lds, acc0, acc1, lh, r);

    #pragma unroll
    for (int t = 0; t < 8; ++t) {
        const int d    = nB + t * 16 + r;
        const float sc = rsqrtf(var[d] + BN_EPS);
        const float sh = (bias[d] - mean[d]) * sc + beta[d];
        #pragma unroll
        for (int g = 0; g < 2; ++g) {
            const v8f* acc = g ? acc1 : acc0;
            const int m0 = mW + g * 16 + 8 * lh;
            #pragma unroll
            for (int j = 0; j < 8; ++j) {
                int m = m0 + j;
                int n = m / HW;
                int p = m - n * HW;
                float v = acc[t][j] * sc + sh;
                v = v > 0.f ? v : 0.f;
                h1[(size_t)(n * CCH + d) * HWP + p] = (h16)v;
            }
        }
    }
}

// ---------------------------------------------------------------------------
// GEMM2: h2(n*HW+q, c) = relu(h1(n*C+c, p) @ wd(p,q) + db(q))
// ---------------------------------------------------------------------------
__global__ __launch_bounds__(256)
void gemm2_kernel(const h16* __restrict__ h1, const h16* __restrict__ wdt,
                  const float* __restrict__ db, h16* __restrict__ h2)
{
    __shared__ h16 lds[2 * PANEL];
    const int wv = threadIdx.x >> 5, lane = threadIdx.x & 31;
    const int lh = lane >> 4, r = lane & 15;
    const int mW = blockIdx.x * 256 + wv * 32;
    const int nB = blockIdx.y * 128;

    v8f acc0[8] = {}, acc1[8] = {};
    gemm_core<3>(h1 + (size_t)(mW + r) * HWP, h1 + (size_t)(mW + 16 + r) * HWP,
                 wdt, HWP, nB, lds, acc0, acc1, lh, r);

    #pragma unroll
    for (int t = 0; t < 8; ++t) {
        const int q = nB + t * 16 + r;
        if (q < HW) {
            const float bq = db[q];
            #pragma unroll
            for (int g = 0; g < 2; ++g) {
                const v8f* acc = g ? acc1 : acc0;
                const int m0 = mW + g * 16 + 8 * lh;   // row = n*256 + c
                const int n  = m0 >> 8;
                const int c0 = m0 & 255;               // 8-aligned within n
                v8h pk;
                #pragma unroll
                for (int j = 0; j < 8; ++j) {
                    float v = acc[t][j] + bq;
                    v = v > 0.f ? v : 0.f;
                    pk[j] = (h16)v;
                }
                *(v8h*)(h2 + (size_t)(n * HW + q) * CCH + c0) = pk;
            }
        }
    }
}

// ---------------------------------------------------------------------------
// GEMM3: out(m, d) = x(m, d) + relu(BN(h2(m, c) @ w2(c, d) + b2))
// ---------------------------------------------------------------------------
__global__ __launch_bounds__(256)
void gemm3_kernel(const h16* __restrict__ h2, const h16* __restrict__ w2t,
                  const float* __restrict__ bias, const float* __restrict__ mean,
                  const float* __restrict__ var,  const float* __restrict__ beta,
                  const float* __restrict__ x, float* __restrict__ out)
{
    __shared__ h16 lds[2 * PANEL];
    const int wv = threadIdx.x >> 5, lane = threadIdx.x & 31;
    const int lh = lane >> 4, r = lane & 15;
    const int mW = blockIdx.x * 256 + wv * 32;
    const int nB = blockIdx.y * 128;

    v8f acc0[8] = {}, acc1[8] = {};
    gemm_core<2>(h2 + (size_t)(mW + r) * CCH, h2 + (size_t)(mW + 16 + r) * CCH,
                 w2t, CCH, nB, lds, acc0, acc1, lh, r);

    #pragma unroll
    for (int t = 0; t < 8; ++t) {
        const int d    = nB + t * 16 + r;
        const float sc = rsqrtf(var[d] + BN_EPS);
        const float sh = (bias[d] - mean[d]) * sc + beta[d];
        #pragma unroll
        for (int g = 0; g < 2; ++g) {
            const v8f* acc = g ? acc1 : acc0;
            const int m0 = mW + g * 16 + 8 * lh;
            #pragma unroll
            for (int j = 0; j < 8; ++j) {
                float v = acc[t][j] * sc + sh;
                v = v > 0.f ? v : 0.f;
                size_t idx = (size_t)(m0 + j) * CCH + d;
                out[idx] = x[idx] + v;
            }
        }
    }
}

// ---------------------------------------------------------------------------
extern "C" void kernel_launch(void* const* d_in, const int* in_sizes, int n_in,
                              void* d_out, int out_size, void* d_ws, size_t ws_size,
                              hipStream_t stream)
{
    const float* x   = (const float*)d_in[0];
    const float* w1  = (const float*)d_in[1];
    const float* b1  = (const float*)d_in[2];
    const float* mu1 = (const float*)d_in[3];
    const float* vr1 = (const float*)d_in[4];
    const float* be1 = (const float*)d_in[5];
    const float* wd  = (const float*)d_in[6];
    const float* db  = (const float*)d_in[7];
    const float* w2  = (const float*)d_in[8];
    const float* b2  = (const float*)d_in[9];
    const float* mu2 = (const float*)d_in[10];
    const float* vr2 = (const float*)d_in[11];
    const float* be2 = (const float*)d_in[12];
    float* out = (float*)d_out;

    char* ws = (char*)d_ws;
    size_t off = 0;
    h16* xh  = (h16*)(ws + off); off += (size_t)NHW * CCH * sizeof(h16); // 47.3 MB
    h16* w1t = (h16*)(ws + off); off += (size_t)CCH * CCH * sizeof(h16);
    h16* w2t = (h16*)(ws + off); off += (size_t)CCH * CCH * sizeof(h16);
    h16* wdt = (h16*)(ws + off); off += (size_t)HWP * HWP * sizeof(h16);
    h16* h1  = (h16*)(ws + off); off += (size_t)M2  * HWP * sizeof(h16); // 50.3 MB
    h16* h2  = (h16*)(ws + off); off += (size_t)NHW * CCH * sizeof(h16); // 47.3 MB

    const long long prep_total = (long long)NHW * CCH
                               + 2LL * CCH * CCH
                               + (long long)HWP * HWP
                               + (long long)M2 * (HWP - HW);
    const int prep_blocks = (int)((prep_total + 255) / 256);
    prep_kernel<<<prep_blocks, 256, 0, stream>>>(x, w1, w2, wd, xh, w1t, w2t, wdt, h1);

    gemm1_kernel<<<dim3(NHW / 256, CCH / 128), 256, 0, stream>>>(
        xh, w1t, b1, mu1, vr1, be1, h1);
    gemm2_kernel<<<dim3(M2 / 256, HWP / 128), 256, 0, stream>>>(
        h1, wdt, db, h2);
    gemm3_kernel<<<dim3(NHW / 256, CCH / 128), 256, 0, stream>>>(
        h2, w2t, b2, mu2, vr2, be2, x, out);
}
